// FastVNN_31817117729490
// MI455X (gfx1250) — matrix operand order repaired and checked by reading, not compiled, not run on hardware
//
#include <hip/hip_runtime.h>
#include <hip/hip_bf16.h>
#include <math.h>

#define GENE_DIM 20000
#define FEATS    8
#define HIDDEN   4
#define BATCH    128
#define HTOT     6000
#define BN_EPS   1e-5f

typedef __attribute__((ext_vector_type(16))) _Float16 v16h;
typedef __attribute__((ext_vector_type(8)))  float    v8f;

__device__ __forceinline__ int imin(int a, int b) { return a < b ? a : b; }

// ---------------------------------------------------------------------------
// Kernel 1: per-gene dot + tanh, coalesced float4 pass over x (82 MB, the
// bandwidth-dominant stage), plus partial sums for batchnorm.
// Thread i = b*GENE_DIM + g  -> x reads and gn writes both fully coalesced.
// ---------------------------------------------------------------------------
__global__ void __launch_bounds__(256)
gene_layer_kernel(const float* __restrict__ x,
                  const float* __restrict__ Wg,
                  const float* __restrict__ bg,
                  float* __restrict__ gn,
                  float* __restrict__ acc)
{
    __shared__ float s_sum[256];
    __shared__ float s_sq[256];

    int i = blockIdx.x * blockDim.x + threadIdx.x;   // i = b*GENE_DIM + g
    int g = i % GENE_DIM;

    const float4* xp = (const float4*)x  + (size_t)i * 2;
    const float4* wp = (const float4*)Wg + (size_t)g * 2;
    float4 a0 = xp[0], a1 = xp[1];
    float4 w0 = wp[0], w1 = wp[1];

    float d = a0.x*w0.x + a0.y*w0.y + a0.z*w0.z + a0.w*w0.w
            + a1.x*w1.x + a1.y*w1.y + a1.z*w1.z + a1.w*w1.w
            + bg[g];
    float t = tanhf(d);
    gn[i] = t;

    s_sum[threadIdx.x] = t;
    s_sq[threadIdx.x]  = t * t;
    __syncthreads();
    for (int s = 128; s > 0; s >>= 1) {
        if (threadIdx.x < s) {
            s_sum[threadIdx.x] += s_sum[threadIdx.x + s];
            s_sq[threadIdx.x]  += s_sq[threadIdx.x + s];
        }
        __syncthreads();
    }
    if (threadIdx.x == 0) {
        atomicAdd(&acc[0], s_sum[0]);
        atomicAdd(&acc[1], s_sq[0]);
    }
}

// ---------------------------------------------------------------------------
// Kernel 2: fold batchnorm into an affine (scale, shift) pair.
// ---------------------------------------------------------------------------
__global__ void bn_finalize_kernel(const float* __restrict__ acc,
                                   const float* __restrict__ gamma,
                                   const float* __restrict__ beta,
                                   float* __restrict__ affine)
{
    float n    = (float)(BATCH * GENE_DIM);
    float mu   = acc[0] / n;
    float var  = acc[1] / n - mu * mu;
    float rstd = rsqrtf(var + BN_EPS);
    float sc   = gamma[0] * rstd;
    affine[0]  = sc;
    affine[1]  = beta[0] - mu * sc;
}

// ---------------------------------------------------------------------------
// Kernel 3: generic graph level as a gather-WMMA, K is compile-time.
// One block = 256 threads = 8 waves; block -> one node/term, wave -> one
// 16-batch tile. A (16xK gathered inputs) and B (Kx4 weights) are built in
// the CDNA5 16-bit WMMA VGPR layouts with *branch-free* clamp+select
// predication, so all gathers issue as one clause with a single wait
// (instead of 16 serialized exec-masked load/wait pairs). One
// v_wmma_f32_16x16x32_f16, then tanh -> dot(w2) via a 16-lane __shfl_xor
// butterfly -> tanh -> write xr[b*H + node].
// Levels feed strictly forward (e_in < node_start), so xr in == xr out.
// ---------------------------------------------------------------------------
template <int K>
__global__ void __launch_bounds__(256)
graph_level_wmma_kernel(const float* __restrict__ src, int src_stride,
                        const int*   __restrict__ e_in,
                        const float* __restrict__ w1,
                        const float* __restrict__ w2,
                        const float* __restrict__ affine,   // null => identity
                        float* __restrict__ xr, int node_start)
{
    int nl   = blockIdx.x;
    int node = node_start + nl;
    int lane = threadIdx.x & 31;
    int wv   = threadIdx.x >> 5;
    int b0   = wv * 16;
    int hf   = lane >> 4;     // which 16-lane half
    int l16  = lane & 15;

    float sc = 1.f, sh = 0.f;
    if (affine) { sc = affine[0]; sh = affine[1]; }

    const int*   eptr = e_in + nl * K;
    const float* w1p  = w1   + nl * K * HIDDEN;
    const float* rowp = src  + (size_t)(b0 + l16) * src_stride;

    // A matrix, 16x32 f16 (M=batch, K=edge). 16-bit A layout:
    // lanes 0-15 : elems 0..7 -> K 0..7,  elems 8..15 -> K 16..23
    // lanes 16-31: elems 0..7 -> K 8..15, elems 8..15 -> K 24..31
    // k = kmin + hf*8 with kmin = e + (e<8 ? 0 : 8).
    v16h amat;
#pragma unroll
    for (int e = 0; e < 16; ++e) {
        const int kmin = e + ((e < 8) ? 0 : 8);
        if (kmin >= K) {                 // dead for both halves: compile-time 0
            amat[e] = (_Float16)0.f;
        } else {
            int  k    = kmin + hf * 8;   // runtime (depends on lane half)
            int  kk   = imin(k, K - 1);  // clamped, always-valid gather
            int  idx  = eptr[kk];
            float v   = sc * rowp[idx] + sh;
            amat[e]   = (_Float16)((k < K) ? v : 0.f);   // v_cndmask, no branch
        }
    }

    // B matrix, 32x16 f16 (K x N), same K order as A: lanes 0-15 hold K 0..7,16..23 (col N = lane),
    // lanes 16-31 hold K 8..15,24..31. Only N < HIDDEN is non-zero.
    v16h bmat;
#pragma unroll
    for (int e = 0; e < 16; ++e) {
        const int kmin = e + ((e < 8) ? 0 : 8);
        if (kmin >= K) {                 // dead for both halves: compile-time 0
            bmat[e] = (_Float16)0.f;
        } else {
            int  k    = kmin + hf * 8;
            int  kk   = imin(k, K - 1);
            float v   = w1p[kk * HIDDEN + (l16 & (HIDDEN - 1))];
            bool ok   = (k < K) && (l16 < HIDDEN);
            bmat[e]   = (_Float16)(ok ? v : 0.f);
        }
    }

    v8f c = {};
    c = __builtin_amdgcn_wmma_f32_16x16x32_f16(false, amat, false, bmat,
                                               (short)0, c, false, false);

    // D layout: VGPR vi -> M = vi + hf*8, N = l16. Hidden channel = N (<4).
    // Branch-free: lanes with N>=4 have c==0 and w2v==0.
    float w2v = w2[nl * HIDDEN + (l16 & (HIDDEN - 1))];
    w2v = (l16 < HIDDEN) ? w2v : 0.f;
#pragma unroll
    for (int vi = 0; vi < 8; ++vi) {
        float hval = tanhf(c[vi]) * w2v;
        hval += __shfl_xor(hval, 1, 32);
        hval += __shfl_xor(hval, 2, 32);
        hval += __shfl_xor(hval, 4, 32);
        hval += __shfl_xor(hval, 8, 32);   // sums lanes within each 16-half
        if (l16 == 0) {
            int b = b0 + vi + hf * 8;
            xr[(size_t)b * HTOT + node] = tanhf(hval);
        }
    }
}

// ---------------------------------------------------------------------------
// Kernel 4: root node (level 4, single node) + head: hidden -> y, logits,
// sigmoid. 128 threads, one per batch row.
// ---------------------------------------------------------------------------
__global__ void root_kernel(const int*   __restrict__ e_in4,
                            const float* __restrict__ w1_4,
                            const float* __restrict__ w2_4,
                            const float* __restrict__ Wa,
                            const float* __restrict__ ba,
                            float* __restrict__ out_sig,
                            float* __restrict__ out_logit,
                            float* __restrict__ xr)
{
    int b = threadIdx.x;
    float h[HIDDEN] = {0.f, 0.f, 0.f, 0.f};
    for (int j = 0; j < 6; ++j) {
        float v = xr[(size_t)b * HTOT + e_in4[j]];
        for (int hh = 0; hh < HIDDEN; ++hh) h[hh] += v * w1_4[j * HIDDEN + hh];
    }
    float y = 0.f, lg = 0.f;
    for (int hh = 0; hh < HIDDEN; ++hh) {
        float t = tanhf(h[hh]);
        y  += t * w2_4[hh];
        lg += t * Wa[hh];
    }
    lg += ba[0];
    xr[(size_t)b * HTOT + (HTOT - 1)] = tanhf(y);
    out_logit[b] = lg;
    out_sig[b]   = 1.f / (1.f + expf(-lg));
}

// ---------------------------------------------------------------------------
extern "C" void kernel_launch(void* const* d_in, const int* in_sizes, int n_in,
                              void* d_out, int out_size, void* d_ws, size_t ws_size,
                              hipStream_t stream)
{
    (void)in_sizes; (void)n_in; (void)out_size; (void)ws_size;

    // setup_inputs() dict order
    const int*   tg_genes = (const int*)  d_in[1];
    const int*   e_in1    = (const int*)  d_in[5];
    const int*   e_in2    = (const int*)  d_in[8];
    const int*   e_in3    = (const int*)  d_in[11];
    const int*   e_in4    = (const int*)  d_in[14];
    const float* x        = (const float*)d_in[15];
    const float* Wg       = (const float*)d_in[16];
    const float* bg       = (const float*)d_in[17];
    const float* gamma    = (const float*)d_in[18];
    const float* beta     = (const float*)d_in[19];
    const float* w1_0     = (const float*)d_in[20];
    const float* w2_0     = (const float*)d_in[21];
    const float* w1_1     = (const float*)d_in[22];
    const float* w2_1     = (const float*)d_in[23];
    const float* w1_2     = (const float*)d_in[24];
    const float* w2_2     = (const float*)d_in[25];
    const float* w1_3     = (const float*)d_in[26];
    const float* w2_3     = (const float*)d_in[27];
    const float* w1_4     = (const float*)d_in[28];
    const float* w2_4     = (const float*)d_in[29];
    const float* Wa       = (const float*)d_in[30];
    const float* ba       = (const float*)d_in[31];

    float* out       = (float*)d_out;
    float* out_sig   = out;                 // (B,1)
    float* out_logit = out + BATCH;         // (B,1)
    float* xr        = out + 2 * BATCH;     // (B,H) row-major, written in place

    char*  ws     = (char*)d_ws;
    float* acc    = (float*)ws;             // 2 floats: sum, sumsq
    float* affine = (float*)(ws + 8);       // 2 floats: scale, shift
    float* gn     = (float*)(ws + 256);     // B*GENE_DIM floats (10.24 MB)

    hipMemsetAsync(acc, 0, 2 * sizeof(float), stream);

    gene_layer_kernel<<<(BATCH * GENE_DIM) / 256, 256, 0, stream>>>(
        x, Wg, bg, gn, acc);

    bn_finalize_kernel<<<1, 1, 0, stream>>>(acc, gamma, beta, affine);

    // Level 0: 4000 terms, K=30 gathered genes, affine-normalized input.
    graph_level_wmma_kernel<30><<<4000, 256, 0, stream>>>(
        gn, GENE_DIM, tg_genes, w1_0, w2_0, affine, xr, 0);
    // Levels 1-3: K=6 predecessors out of xr (identity affine).
    graph_level_wmma_kernel<6><<<1400, 256, 0, stream>>>(
        xr, HTOT, e_in1, w1_1, w2_1, nullptr, xr, 4000);
    graph_level_wmma_kernel<6><<<550, 256, 0, stream>>>(
        xr, HTOT, e_in2, w1_2, w2_2, nullptr, xr, 5400);
    graph_level_wmma_kernel<6><<<49, 256, 0, stream>>>(
        xr, HTOT, e_in3, w1_3, w2_3, nullptr, xr, 5950);

    // Level 4 (root) + classification head.
    root_kernel<<<1, 128, 0, stream>>>(
        e_in4, w1_4, w2_4, Wa, ba, out_sig, out_logit, xr);
}
